// CanonCausalMultiheadAttn_51642686767713
// MI455X (gfx1250) — matrix operand order repaired and checked by reading, not compiled, not run on hardware
//
#include <hip/hip_runtime.h>
#include <hip/hip_bf16.h>

typedef __bf16 bh;
typedef __attribute__((ext_vector_type(16))) __bf16 v16bf;
typedef __attribute__((ext_vector_type(8)))  __bf16 v8bf;
typedef __attribute__((ext_vector_type(8)))  float  v8f;

#define DEV static __device__ __forceinline__

// ---------------------------------------------------------------------------
// WMMA fragment helpers (CDNA5 16x16x32 bf16, wave32).
// A (16x32, MxK), row-major in memory with leading dim ld:
//   lane<16 : row=lane,    K = {0..7} in f[0..7],  {16..23} in f[8..15]
//   lane>=16: row=lane-16, K = {8..15},            {24..31}
// ---------------------------------------------------------------------------
DEV v16bf load_a_frag(const bh* base, int ld) {
  int lane = threadIdx.x & 31;
  int r    = lane & 15;
  int kb   = (lane >> 4) << 3;            // 0 or 8
  const bh* p = base + (size_t)r * ld + kb;
  v8bf lo = *(const v8bf*)(p);
  v8bf hi = *(const v8bf*)(p + 16);
  v16bf f;
#pragma unroll
  for (int i = 0; i < 8; ++i) { f[i] = lo[i]; f[8 + i] = hi[i]; }
  return f;
}

// B (32x16, KxN). Memory holds Bt[N][K] row-major (column n of B contiguous).
//   lane<16 : col=lane,    K = {0..7} in f[0..7],  {16..23} in f[8..15]
//   lane>=16: col=lane-16, K = {8..15},            {24..31}
DEV v16bf load_b_frag(const bh* base, int ld) {
  int lane = threadIdx.x & 31;
  int n    = lane & 15;
  int kb   = (lane >> 4) << 3;            // 0 or 8
  const bh* p = base + (size_t)n * ld + kb;
  v8bf lo = *(const v8bf*)(p);
  v8bf hi = *(const v8bf*)(p + 16);
  v16bf f;
#pragma unroll
  for (int i = 0; i < 8; ++i) { f[i] = lo[i]; f[8 + i] = hi[i]; }
  return f;
}

DEV v8f wmma_bf16(v16bf a, v16bf b, v8f c) {
  return __builtin_amdgcn_wmma_f32_16x16x32_bf16(false, a, false, b,
                                                 (short)0, c, false, false);
}

// ---------------------------------------------------------------------------
// fp32 -> bf16 convert
// ---------------------------------------------------------------------------
__global__ void k_cvt(const float* __restrict__ src, bh* __restrict__ dst, int n) {
  int i = blockIdx.x * 256 + threadIdx.x;
  if (i < n) dst[i] = (bh)src[i];
}

// ---------------------------------------------------------------------------
// C[M,N] (fp32) = A[M,K] (bf16) * Wt[N,K]^T (bf16).  64x64 tile per wave.
// ---------------------------------------------------------------------------
__global__ __launch_bounds__(128)
void k_gemm(const bh* __restrict__ A, const bh* __restrict__ Wt,
            float* __restrict__ C, int M, int N, int K) {
  int wave = threadIdx.x >> 5;
  int lane = threadIdx.x & 31;
  int n0 = blockIdx.x * 64;
  int m0 = (blockIdx.y * 4 + wave) * 64;
  if (m0 >= M || n0 >= N) return;

  v8f acc[4][4];
#pragma unroll
  for (int i = 0; i < 4; ++i)
#pragma unroll
    for (int j = 0; j < 4; ++j) acc[i][j] = (v8f)0.0f;

  for (int k0 = 0; k0 < K; k0 += 32) {
    if (k0 + 32 < K) {   // hint next K-block (global_prefetch_b8)
      __builtin_prefetch(A  + (size_t)m0 * K + k0 + 32, 0, 1);
      __builtin_prefetch(Wt + (size_t)n0 * K + k0 + 32, 0, 1);
    }
    v16bf af[4], bf_[4];
#pragma unroll
    for (int i = 0; i < 4; ++i)
      af[i] = load_a_frag(A + (size_t)(m0 + i * 16) * K + k0, K);
#pragma unroll
    for (int j = 0; j < 4; ++j)
      bf_[j] = load_b_frag(Wt + (size_t)(n0 + j * 16) * K + k0, K);
#pragma unroll
    for (int i = 0; i < 4; ++i)
#pragma unroll
      for (int j = 0; j < 4; ++j)
        acc[i][j] = wmma_bf16(af[i], bf_[j], acc[i][j]);
  }

  int nn = lane & 15;
  int mh = (lane >> 4) * 8;
#pragma unroll
  for (int i = 0; i < 4; ++i)
#pragma unroll
    for (int j = 0; j < 4; ++j)
#pragma unroll
      for (int r = 0; r < 8; ++r)
        C[(size_t)(m0 + i * 16 + mh + r) * N + n0 + j * 16 + nn] = acc[i][j][r];
}

// ---------------------------------------------------------------------------
// Canon-B conv + residual, per-head RMSNorm, RoPE.  One block = (b,t,head).
// blockIdx.y: 0..15 = q-heads, 16..23 = k-heads, 24..31 = v-heads.
// Writes Q [B,16,S,128] bf16, K [B,8,S,128] bf16, V transposed [B,8,128,S] bf16.
// ---------------------------------------------------------------------------
__global__ __launch_bounds__(128)
void k_canon(const float* __restrict__ qf, const float* __restrict__ kf,
             const float* __restrict__ vf,
             const float* __restrict__ wqc, const float* __restrict__ wkc,
             const float* __restrict__ wvc,
             const float* __restrict__ qn, const float* __restrict__ kn,
             bh* __restrict__ qo, bh* __restrict__ ko, bh* __restrict__ vto) {
  const int S = 2048;
  int bt = blockIdx.x;
  int b = bt >> 11, t = bt & 2047;
  int hh = blockIdx.y;
  int d  = threadIdx.x;

  const float* src; const float* wc; int Cdim, c;
  if (hh < 16)      { src = qf; wc = wqc; Cdim = 2048; c = hh * 128 + d; }
  else if (hh < 24) { src = kf; wc = wkc; Cdim = 1024; c = (hh - 16) * 128 + d; }
  else              { src = vf; wc = wvc; Cdim = 1024; c = (hh - 24) * 128 + d; }

  float xt = src[(size_t)bt * Cdim + c];
  float conv = 0.0f;
#pragma unroll
  for (int kk = 0; kk < 4; ++kk) {
    int tt = t - kk;
    float xv = (tt >= 0) ? src[(size_t)(b * S + tt) * Cdim + c] : 0.0f;
    conv = fmaf(wc[c * 4 + kk], xv, conv);
  }
  float y = xt + conv;

  if (hh < 24) {
    // RMS norm across the 128 channels of this head (4 wave32's)
    __shared__ float wsum[4];
    __shared__ float vals[128];
    float ss = y * y;
#pragma unroll
    for (int off = 16; off >= 1; off >>= 1) ss += __shfl_xor(ss, off, 32);
    if ((threadIdx.x & 31) == 0) wsum[threadIdx.x >> 5] = ss;
    __syncthreads();
    float tot = wsum[0] + wsum[1] + wsum[2] + wsum[3];
    float inv = rsqrtf(tot * (1.0f / 128.0f) + 1e-6f);
    float yn = y * inv * (hh < 16 ? qn[d] : kn[d]);
    vals[d] = yn;
    __syncthreads();
    float partner = vals[d ^ 64];
    int j = d & 63;
    // inv_freq = 10000^(-j/64) = exp(-j*ln(1e4)/64)
    float invf = __expf(-(float)j * 0.14391156f);
    float ang = (float)t * invf;
    float sn, cs;
    __sincosf(ang, &sn, &cs);
    float out = (d < 64) ? (yn * cs - partner * sn) : (yn * cs + partner * sn);
    if (hh < 16) qo[((size_t)(b * 16 + hh) * S + t) * 128 + d] = (bh)out;
    else         ko[((size_t)(b * 8 + (hh - 16)) * S + t) * 128 + d] = (bh)out;
  } else {
    int hv = hh - 24;
    vto[((size_t)(b * 8 + hv) * 128 + d) * S + t] = (bh)y;
  }
}

// ---------------------------------------------------------------------------
// Flash-style causal attention. 8 waves/block, each wave = one 16-row Q tile.
// Q [B,16,S,128], K [B,8,S,128], Vt [B,8,128,S] -> O [B,S,16,128] (bf16).
// ---------------------------------------------------------------------------
__global__ __launch_bounds__(256)
void k_attn(const bh* __restrict__ Q, const bh* __restrict__ Kb,
            const bh* __restrict__ Vt, bh* __restrict__ O) {
  const int S = 2048;
  int bhd = blockIdx.y;                 // b*16+h
  int b = bhd >> 4, h = bhd & 15;
  int hk = h >> 1;                      // GQA: groups = 2
  int wave = threadIdx.x >> 5;
  int lane = threadIdx.x & 31;
  int q0 = blockIdx.x * 128 + wave * 16;

  const bh* qbase = Q  + (size_t)(b * 16 + h)  * S * 128;
  const bh* kbase = Kb + (size_t)(b * 8 + hk)  * S * 128;
  const bh* vbase = Vt + (size_t)(b * 8 + hk)  * 128 * S;

  __shared__ __align__(16) bh plds[8][16 * 32];
  bh* myp = &plds[wave][0];

  v16bf aq[4];
#pragma unroll
  for (int i = 0; i < 4; ++i)
    aq[i] = load_a_frag(qbase + (size_t)q0 * 128 + i * 32, 128);

  v8f acc[8];
#pragma unroll
  for (int jd = 0; jd < 8; ++jd) acc[jd] = (v8f)0.0f;
  float mrun[8], lrun[8];
#pragma unroll
  for (int j = 0; j < 8; ++j) { mrun[j] = -1e30f; lrun[j] = 0.0f; }

  const float scale = 0.08838834764831845f;   // 1/sqrt(128)
  int nn = lane & 15;
  int mh = (lane >> 4) * 8;

  for (int n0 = 0; n0 < q0 + 16; n0 += 32) {
    // ---- scores: two 16x16 tiles over d=128 (4 WMMAs each) ----
    v8f st0 = (v8f)0.0f, st1 = (v8f)0.0f;
#pragma unroll
    for (int dc = 0; dc < 4; ++dc) {
      v16bf bk0 = load_b_frag(kbase + (size_t)n0 * 128 + dc * 32, 128);
      st0 = wmma_bf16(aq[dc], bk0, st0);
    }
#pragma unroll
    for (int dc = 0; dc < 4; ++dc) {
      v16bf bk1 = load_b_frag(kbase + (size_t)(n0 + 16) * 128 + dc * 32, 128);
      st1 = wmma_bf16(aq[dc], bk1, st1);
    }
    // ---- online softmax (rows = (mh+j); 16-lane groups stay disjoint) ----
    float p0[8], p1[8];
#pragma unroll
    for (int j = 0; j < 8; ++j) {
      int m = q0 + mh + j;
      float s0 = (n0 + nn      <= m) ? st0[j] * scale : -1e30f;
      float s1 = (n0 + 16 + nn <= m) ? st1[j] * scale : -1e30f;
      float rmax = fmaxf(s0, s1);
#pragma unroll
      for (int off = 8; off >= 1; off >>= 1)
        rmax = fmaxf(rmax, __shfl_xor(rmax, off, 32));
      float newm = fmaxf(mrun[j], rmax);
      float e0 = __expf(s0 - newm);
      float e1 = __expf(s1 - newm);
      float rs = e0 + e1;
#pragma unroll
      for (int off = 8; off >= 1; off >>= 1) rs += __shfl_xor(rs, off, 32);
      float corr = __expf(mrun[j] - newm);
      lrun[j] = lrun[j] * corr + rs;
      mrun[j] = newm;
#pragma unroll
      for (int jd = 0; jd < 8; ++jd) acc[jd][j] *= corr;
      p0[j] = e0; p1[j] = e1;
    }
    // ---- transpose P (C-layout -> A-fragment) through LDS ----
#pragma unroll
    for (int j = 0; j < 8; ++j) {
      myp[(mh + j) * 32 + nn]      = (bh)p0[j];
      myp[(mh + j) * 32 + 16 + nn] = (bh)p1[j];
    }
    asm volatile("s_wait_dscnt 0" ::: "memory");
    v16bf pa = load_a_frag(myp, 32);
    // ---- acc += P (16x32) @ V (32x128) ----
#pragma unroll
    for (int jd = 0; jd < 8; ++jd) {
      v16bf bv = load_b_frag(vbase + (size_t)(jd * 16) * S + n0, S);
      acc[jd] = wmma_bf16(pa, bv, acc[jd]);
    }
  }

  // ---- epilogue: normalize and write bf16 attn output [B,S,16,128] ----
#pragma unroll
  for (int j = 0; j < 8; ++j) {
    float invl = 1.0f / lrun[j];
    int srow = q0 + mh + j;
#pragma unroll
    for (int jd = 0; jd < 8; ++jd) {
      float o = acc[jd][j] * invl;
      O[((size_t)(b * S + srow) * 16 + h) * 128 + jd * 16 + nn] = (bh)o;
    }
  }
}

// ---------------------------------------------------------------------------
extern "C" void kernel_launch(void* const* d_in, const int* in_sizes, int n_in,
                              void* d_out, int out_size, void* d_ws, size_t ws_size,
                              hipStream_t stream) {
  const int B = 2, S = 2048, D = 2048, Dkv = 1024, H = 16;
  const size_t MS = (size_t)B * S;                // 4096 rows

  const float* hs  = (const float*)d_in[0];
  const float* Wq  = (const float*)d_in[1];
  const float* Wk  = (const float*)d_in[2];
  const float* Wv  = (const float*)d_in[3];
  const float* Wo  = (const float*)d_in[4];
  const float* cqw = (const float*)d_in[5];
  const float* ckw = (const float*)d_in[6];
  const float* cvw = (const float*)d_in[7];
  const float* qnw = (const float*)d_in[8];
  const float* knw = (const float*)d_in[9];

  char* w = (char*)d_ws;
  auto carve = [&](size_t bytes) {
    char* p = w;
    w += (bytes + 255) & ~(size_t)255;
    return (void*)p;
  };
  bh* hs_bf = (bh*)carve(MS * D * 2);
  bh* wq_bf = (bh*)carve((size_t)D * D * 2);
  bh* wk_bf = (bh*)carve((size_t)Dkv * D * 2);
  bh* wv_bf = (bh*)carve((size_t)Dkv * D * 2);
  bh* wo_bf = (bh*)carve((size_t)D * D * 2);
  float* qf = (float*)carve(MS * D * 4);
  float* kf = (float*)carve(MS * Dkv * 4);
  float* vf = (float*)carve(MS * Dkv * 4);
  bh* qb = (bh*)carve(MS * D * 2);     // [B,16,S,128]
  bh* kb = (bh*)carve(MS * Dkv * 2);   // [B,8,S,128]
  bh* vt = (bh*)carve(MS * Dkv * 2);   // [B,8,128,S]
  bh* ab = (bh*)carve(MS * D * 2);     // attn out [B,S,16,128]

  auto cvt = [&](const float* s, bh* dst, size_t n) {
    k_cvt<<<dim3((unsigned)((n + 255) / 256)), 256, 0, stream>>>(s, dst, (int)n);
  };
  cvt(hs, hs_bf, MS * D);
  cvt(Wq, wq_bf, (size_t)D * D);
  cvt(Wk, wk_bf, (size_t)Dkv * D);
  cvt(Wv, wv_bf, (size_t)Dkv * D);
  cvt(Wo, wo_bf, (size_t)D * D);

  // QKV projections
  k_gemm<<<dim3(D / 64,   MS / 256), 128, 0, stream>>>(hs_bf, wq_bf, qf, (int)MS, D,   D);
  k_gemm<<<dim3(Dkv / 64, MS / 256), 128, 0, stream>>>(hs_bf, wk_bf, kf, (int)MS, Dkv, D);
  k_gemm<<<dim3(Dkv / 64, MS / 256), 128, 0, stream>>>(hs_bf, wv_bf, vf, (int)MS, Dkv, D);

  // conv + norm + rope
  k_canon<<<dim3((unsigned)MS, 32), 128, 0, stream>>>(qf, kf, vf, cqw, ckw, cvw,
                                                      qnw, knw, qb, kb, vt);
  // causal attention
  k_attn<<<dim3(S / 128, B * H), 256, 0, stream>>>(qb, kb, vt, ab);

  // output projection -> fp32 d_out
  k_gemm<<<dim3(D / 64, MS / 256), 128, 0, stream>>>(ab, wo_bf, (float*)d_out,
                                                     (int)MS, D, D);
}
